// MPNNEncoder_56324201120052
// MI455X (gfx1250) — compile-verified, matrix operand order checked by reading, not run on hardware
//
#include <hip/hip_runtime.h>
#include <cstdint>
#include <cstddef>

// ---------------------------------------------------------------------------
// MPNN encoder (ProteinMPNN-style) for MI455X / gfx1250, wave32 + WMMA bf16.
// B=2, N=2048, K=48, H=128, L=3.  ~2e11 FLOPs -> compute bound -> bf16 WMMA
// with f32 accumulation; f32 master copies for residual/LayerNorm accuracy.
// ---------------------------------------------------------------------------

typedef __attribute__((ext_vector_type(16))) __bf16 bf16x16;
typedef __attribute__((ext_vector_type(8)))  __bf16 bf16x8;
typedef __attribute__((ext_vector_type(8)))  float  f32x8;

constexpr int Bc = 2, Nc = 2048, Kc = 48, Hc = 128, Lc = 3;
constexpr float SCALE_INV = 1.0f / 30.0f;
constexpr float LN_EPS = 1e-5f;

__device__ __forceinline__ f32x8 zero8() {
  f32x8 z;
#pragma unroll
  for (int i = 0; i < 8; ++i) z[i] = 0.0f;
  return z;
}

__device__ __forceinline__ float gelu_f(float x) {
  // exact erf GELU (torch nn.GELU default)
  return 0.5f * x * (1.0f + erff(x * 0.70710678118654752440f));
}

__device__ __forceinline__ f32x8 wmma_bf16(bf16x16 a, bf16x16 b, f32x8 c) {
  return __builtin_amdgcn_wmma_f32_16x16x32_bf16(
      /*neg_a=*/false, a, /*neg_b=*/false, b,
      /*c_mod=*/(short)0, c, /*reuse_a=*/false, /*reuse_b=*/false);
}

// A fragment (16x32 bf16, M x K) from row-major LDS matrix.
// Lane l: M = l&15, khalf = l>>4.  The required VGPR layout is exactly two
// 16-byte contiguous runs per lane: K = kt*32+8*khalf+[0..7] and +16+[0..7].
__device__ __forceinline__ bf16x16 load_a_frag(const __bf16* smem, int ldm,
                                               int mtile, int ktile) {
  int lane = threadIdx.x & 31;
  int m = lane & 15;
  int khalf = lane >> 4;
  const __bf16* p = smem + (size_t)(mtile * 16 + m) * ldm + ktile * 32 + khalf * 8;
  bf16x8 lo = *(const bf16x8*)(p);
  bf16x8 hi = *(const bf16x8*)(p + 16);
  bf16x16 r;
#pragma unroll
  for (int i = 0; i < 8; ++i) { r[i] = lo[i]; r[i + 8] = hi[i]; }
  return r;
}

// B fragment (32x16 bf16, K x N) from pre-repacked weights: per fragment,
// per lane, 16 contiguous bf16 (two global b128 loads, L2-resident).
__device__ __forceinline__ bf16x16 load_b_frag(const __bf16* wpack, int ntiles,
                                               int ktile, int ntile) {
  int lane = threadIdx.x & 31;
  const __bf16* p = wpack + (((size_t)(ktile * ntiles + ntile)) * 32 + lane) * 16;
  bf16x8 lo = *(const bf16x8*)(p);
  bf16x8 hi = *(const bf16x8*)(p + 8);
  bf16x16 r;
#pragma unroll
  for (int i = 0; i < 8; ++i) { r[i] = lo[i]; r[i + 8] = hi[i]; }
  return r;
}

// ---------------------------------------------------------------------------
// Weight repack: f32 [Kdim x Ndim] row-major -> bf16 B-fragment order.
// Fragment (kt, nt), lane l (n = nt*16 + (l&15), khalf = l>>4),
// element e -> K = kt*32 + (e&7) + 16*(e>>3) + 8*khalf.
// ---------------------------------------------------------------------------
__global__ __launch_bounds__(256) void repack_weight(const float* __restrict__ W,
                                                     __bf16* __restrict__ P,
                                                     int Kdim, int Ndim) {
  int ntiles = Ndim / 16;
  int total = (Kdim / 32) * ntiles * 32;
  int tid = blockIdx.x * blockDim.x + threadIdx.x;
  if (tid >= total) return;
  int lane = tid & 31;
  int frag = tid >> 5;
  int nt = frag % ntiles;
  int kt = frag / ntiles;
  int n = nt * 16 + (lane & 15);
  int khalf = lane >> 4;
  __bf16* dst = P + (size_t)tid * 16;
#pragma unroll
  for (int e = 0; e < 16; ++e) {
    int krel = (e & 7) + ((e >> 3) * 16) + khalf * 8;
    dst[e] = (__bf16)W[(size_t)(kt * 32 + krel) * Ndim + n];
  }
}

// ---------------------------------------------------------------------------
// h_E = E @ We + be.  Grid: (B*N*K)/128 blocks, 256 threads (8 waves).
// Block does 128 rows x 128 cols; wave w owns M-tile w, all 8 N-tiles.
// ---------------------------------------------------------------------------
__global__ __launch_bounds__(256) void edge_embed(const float* __restrict__ E,
                                                  const __bf16* __restrict__ WeP,
                                                  const float* __restrict__ Web,
                                                  float* __restrict__ hEf,
                                                  __bf16* __restrict__ hEb) {
  __shared__ __attribute__((aligned(16))) __bf16 sA[128 * 128];
  size_t rowbase = (size_t)blockIdx.x * 128;
  const float4* src4 = (const float4*)(E + rowbase * Hc);
  for (int i = threadIdx.x; i < 128 * 128 / 4; i += blockDim.x) {
    float4 v = src4[i];
    sA[i * 4 + 0] = (__bf16)v.x;
    sA[i * 4 + 1] = (__bf16)v.y;
    sA[i * 4 + 2] = (__bf16)v.z;
    sA[i * 4 + 3] = (__bf16)v.w;
  }
  __syncthreads();

  int wave = threadIdx.x >> 5, lane = threadIdx.x & 31;
  f32x8 acc[8];
#pragma unroll
  for (int nt = 0; nt < 8; ++nt) acc[nt] = zero8();

  for (int kt = 0; kt < 4; ++kt) {
    bf16x16 a = load_a_frag(sA, 128, wave, kt);
#pragma unroll
    for (int nt = 0; nt < 8; ++nt)
      acc[nt] = wmma_bf16(a, load_b_frag(WeP, 8, kt, nt), acc[nt]);
  }

  int nlo = lane & 15, mh = lane >> 4;
#pragma unroll
  for (int nt = 0; nt < 8; ++nt) {
    int col = nt * 16 + nlo;
    float bias = Web[col];
#pragma unroll
    for (int v = 0; v < 8; ++v) {
      int m = wave * 16 + v + 8 * mh;
      size_t o = (rowbase + m) * Hc + col;
      float val = acc[nt][v] + bias;
      hEf[o] = val;
      hEb[o] = (__bf16)val;
    }
  }
}

__global__ __launch_bounds__(256) void init_hv(float* hVf, __bf16* hVb, int n) {
  int i = blockIdx.x * blockDim.x + threadIdx.x;
  if (i < n) { hVf[i] = 0.0f; hVb[i] = (__bf16)0.0f; }
}

__global__ __launch_bounds__(256) void make_mask_attend(const float* __restrict__ mask,
                                                        const int* __restrict__ Eidx,
                                                        float* __restrict__ ma) {
  int i = blockIdx.x * blockDim.x + threadIdx.x;
  if (i >= Bc * Nc * Kc) return;
  int bn = i / Kc;
  int b = bn / Nc;
  ma[i] = mask[bn] * mask[b * Nc + Eidx[i]];
}

// GEMM over a 48-row LDS activation, wave's output strip = ntile=wave.
__device__ __forceinline__ void gemm48(const __bf16* A, int lda,
                                       const __bf16* Wp, int wNtiles,
                                       int ktiles, int ntile, f32x8 acc[3]) {
  for (int kt = 0; kt < ktiles; ++kt) {
    bf16x16 bf = load_b_frag(Wp, wNtiles, kt, ntile);
#pragma unroll
    for (int mt = 0; mt < 3; ++mt)
      acc[mt] = wmma_bf16(load_a_frag(A, lda, mt, kt), bf, acc[mt]);
  }
}

// ---------------------------------------------------------------------------
// Node message: one workgroup per (b,n).  Build h_EV[48,384] in LDS,
// 3-layer MLP via WMMA, masked K-reduction, residual + LayerNorm1.
// ---------------------------------------------------------------------------
__global__ __launch_bounds__(256) void node_message(
    const __bf16* __restrict__ hVb, const float* __restrict__ hVf,
    const __bf16* __restrict__ hEb, const int* __restrict__ Eidx,
    const float* __restrict__ maskAtt,
    const __bf16* __restrict__ W1p, const __bf16* __restrict__ W2p,
    const __bf16* __restrict__ W3p,
    const float* __restrict__ b1, const float* __restrict__ b2,
    const float* __restrict__ b3,
    const float* __restrict__ lng, const float* __restrict__ lnb,
    float* __restrict__ hVmid_f, __bf16* __restrict__ hVmid_b) {
  __shared__ __attribute__((aligned(16))) __bf16 sEV[Kc * 384];
  __shared__ __attribute__((aligned(16))) __bf16 sM1[Kc * Hc];
  __shared__ __attribute__((aligned(16))) __bf16 sM2[Kc * Hc];
  __shared__ int sIdx[Kc];
  __shared__ float sMA[Kc];
  __shared__ float sDh[Hc];
  __shared__ float sX[Hc];
  __shared__ float sStat[2];

  int bn = blockIdx.x;
  int b = bn / Nc;
  int tid = threadIdx.x, wave = tid >> 5, lane = tid & 31;

  if (tid < Kc) {
    sIdx[tid] = Eidx[(size_t)bn * Kc + tid];
    sMA[tid] = maskAtt[(size_t)bn * Kc + tid];
  }
  __syncthreads();

  const __bf16* self = hVb + (size_t)bn * Hc;
  const __bf16* he = hEb + (size_t)bn * Kc * Hc;
  for (int i = tid; i < Kc * 16; i += blockDim.x) {
    int k = i >> 4, c = (i & 15) * 8;
    *(bf16x8*)(&sEV[k * 384 + c]) = *(const bf16x8*)(self + c);
    *(bf16x8*)(&sEV[k * 384 + 128 + c]) =
        *(const bf16x8*)(hVb + (size_t)(b * Nc + sIdx[k]) * Hc + c);
    *(bf16x8*)(&sEV[k * 384 + 256 + c]) = *(const bf16x8*)(he + (size_t)k * Hc + c);
  }
  __syncthreads();

  int nlo = lane & 15, mh = lane >> 4, col = wave * 16 + nlo;

  f32x8 acc[3] = {zero8(), zero8(), zero8()};
  gemm48(sEV, 384, W1p, 8, 12, wave, acc);
  {
    float bias = b1[col];
#pragma unroll
    for (int mt = 0; mt < 3; ++mt)
#pragma unroll
      for (int v = 0; v < 8; ++v)
        sM1[(mt * 16 + v + 8 * mh) * Hc + col] = (__bf16)gelu_f(acc[mt][v] + bias);
  }
  __syncthreads();

  f32x8 acc2[3] = {zero8(), zero8(), zero8()};
  gemm48(sM1, Hc, W2p, 8, 4, wave, acc2);
  {
    float bias = b2[col];
#pragma unroll
    for (int mt = 0; mt < 3; ++mt)
#pragma unroll
      for (int v = 0; v < 8; ++v)
        sM2[(mt * 16 + v + 8 * mh) * Hc + col] = (__bf16)gelu_f(acc2[mt][v] + bias);
  }
  __syncthreads();

  f32x8 acc3[3] = {zero8(), zero8(), zero8()};
  gemm48(sM2, Hc, W3p, 8, 4, wave, acc3);
  {
    float bias = b3[col];
    float part = 0.0f;
#pragma unroll
    for (int mt = 0; mt < 3; ++mt)
#pragma unroll
      for (int v = 0; v < 8; ++v) {
        int row = mt * 16 + v + 8 * mh;
        part += sMA[row] * (acc3[mt][v] + bias);
      }
    part += __shfl_xor(part, 16);          // wave32: combine the two M-halves
    if (lane < 16) sDh[col] = part * SCALE_INV;
  }
  __syncthreads();

  if (tid < Hc) sX[tid] = hVf[(size_t)bn * Hc + tid] + sDh[tid];
  __syncthreads();
  if (wave == 0) {
    float s = 0.0f, s2 = 0.0f;
#pragma unroll
    for (int i = 0; i < 4; ++i) {
      float x = sX[lane + 32 * i];
      s += x; s2 += x * x;
    }
#pragma unroll
    for (int off = 16; off; off >>= 1) { s += __shfl_xor(s, off); s2 += __shfl_xor(s2, off); }
    if (lane == 0) {
      float mu = s / Hc;
      sStat[0] = mu;
      sStat[1] = rsqrtf(s2 / Hc - mu * mu + LN_EPS);
    }
  }
  __syncthreads();
  if (tid < Hc) {
    float y = lng[tid] * (sX[tid] - sStat[0]) * sStat[1] + lnb[tid];
    hVmid_f[(size_t)bn * Hc + tid] = y;
    hVmid_b[(size_t)bn * Hc + tid] = (__bf16)y;
  }
}

// ---------------------------------------------------------------------------
// Position-wise FFN: 32 nodes per workgroup.  128->512 GELU ->128,
// residual + LayerNorm2, mask.
// ---------------------------------------------------------------------------
__global__ __launch_bounds__(256) void ffn_kernel(
    const float* __restrict__ hVmid_f, const __bf16* __restrict__ hVmid_b,
    const __bf16* __restrict__ Wip, const float* __restrict__ bi,
    const __bf16* __restrict__ Wop, const float* __restrict__ bo,
    const float* __restrict__ lng, const float* __restrict__ lnb,
    const float* __restrict__ mask,
    float* __restrict__ hVf, __bf16* __restrict__ hVb) {
  __shared__ __attribute__((aligned(16))) __bf16 sX[32 * Hc];
  __shared__ __attribute__((aligned(16))) __bf16 sU[32 * 512];
  int tid = threadIdx.x, wave = tid >> 5, lane = tid & 31;
  size_t row0 = (size_t)blockIdx.x * 32;

  for (int i = tid; i < 32 * 16; i += blockDim.x) {
    int r = i >> 4, c = (i & 15) * 8;
    *(bf16x8*)(&sX[r * Hc + c]) = *(const bf16x8*)(hVmid_b + (row0 + r) * Hc + c);
  }
  __syncthreads();

  int nlo = lane & 15, mh = lane >> 4;

  // GEMM1: [32,128] x [128,512]; wave w owns N-tiles w*4..w*4+3, M-tiles 0..1
  f32x8 acc[2][4];
#pragma unroll
  for (int mt = 0; mt < 2; ++mt)
#pragma unroll
    for (int j = 0; j < 4; ++j) acc[mt][j] = zero8();
  for (int kt = 0; kt < 4; ++kt) {
    bf16x16 a0 = load_a_frag(sX, Hc, 0, kt);
    bf16x16 a1 = load_a_frag(sX, Hc, 1, kt);
#pragma unroll
    for (int j = 0; j < 4; ++j) {
      bf16x16 bf = load_b_frag(Wip, 32, kt, wave * 4 + j);
      acc[0][j] = wmma_bf16(a0, bf, acc[0][j]);
      acc[1][j] = wmma_bf16(a1, bf, acc[1][j]);
    }
  }
#pragma unroll
  for (int j = 0; j < 4; ++j) {
    int col = (wave * 4 + j) * 16 + nlo;
    float bias = bi[col];
#pragma unroll
    for (int mt = 0; mt < 2; ++mt)
#pragma unroll
      for (int v = 0; v < 8; ++v)
        sU[(mt * 16 + v + 8 * mh) * 512 + col] = (__bf16)gelu_f(acc[mt][j][v] + bias);
  }
  __syncthreads();

  // GEMM2: [32,512] x [512,128]; wave w owns N-tile w, M-tiles 0..1
  f32x8 a2[2] = {zero8(), zero8()};
  for (int kt = 0; kt < 16; ++kt) {
    bf16x16 bf = load_b_frag(Wop, 8, kt, wave);
    a2[0] = wmma_bf16(load_a_frag(sU, 512, 0, kt), bf, a2[0]);
    a2[1] = wmma_bf16(load_a_frag(sU, 512, 1, kt), bf, a2[1]);
  }
  __syncthreads();                       // all waves done reading sU
  float* sZ = (float*)sU;                // reuse: 32*128*4 = 16KB <= 32KB
  {
    int col = wave * 16 + nlo;
    float bias = bo[col];
#pragma unroll
    for (int mt = 0; mt < 2; ++mt)
#pragma unroll
      for (int v = 0; v < 8; ++v) {
        int row = mt * 16 + v + 8 * mh;
        float x = hVmid_f[(row0 + row) * Hc + col];
        sZ[row * Hc + col] = x + a2[mt][v] + bias;
      }
  }
  __syncthreads();

  // LayerNorm2 + mask: 4 rows per wave
  for (int r = wave * 4; r < wave * 4 + 4; ++r) {
    size_t node = row0 + r;
    float mk = mask[node];
    float xv[4], s = 0.0f, s2 = 0.0f;
#pragma unroll
    for (int i = 0; i < 4; ++i) {
      float x = sZ[r * Hc + lane + 32 * i];
      xv[i] = x; s += x; s2 += x * x;
    }
#pragma unroll
    for (int off = 16; off; off >>= 1) { s += __shfl_xor(s, off); s2 += __shfl_xor(s2, off); }
    float mu = s / Hc;
    float rs = rsqrtf(s2 / Hc - mu * mu + LN_EPS);
#pragma unroll
    for (int i = 0; i < 4; ++i) {
      int c = lane + 32 * i;
      float y = (lng[c] * (xv[i] - mu) * rs + lnb[c]) * mk;
      hVf[node * Hc + c] = y;
      hVb[node * Hc + c] = (__bf16)y;
    }
  }
}

// ---------------------------------------------------------------------------
// Edge update: same MLP structure; per-edge residual + LayerNorm3, in place.
// ---------------------------------------------------------------------------
__global__ __launch_bounds__(256) void edge_update(
    const __bf16* __restrict__ hVb, float* __restrict__ hEf,
    __bf16* __restrict__ hEb, const int* __restrict__ Eidx,
    const __bf16* __restrict__ W11p, const __bf16* __restrict__ W12p,
    const __bf16* __restrict__ W13p,
    const float* __restrict__ b11, const float* __restrict__ b12,
    const float* __restrict__ b13,
    const float* __restrict__ lng, const float* __restrict__ lnb) {
  __shared__ __attribute__((aligned(16))) __bf16 sEV[Kc * 384];
  __shared__ __attribute__((aligned(16))) __bf16 sM1[Kc * Hc];
  __shared__ __attribute__((aligned(16))) __bf16 sM2[Kc * Hc];
  __shared__ int sIdx[Kc];

  int bn = blockIdx.x;
  int b = bn / Nc;
  int tid = threadIdx.x, wave = tid >> 5, lane = tid & 31;

  if (tid < Kc) sIdx[tid] = Eidx[(size_t)bn * Kc + tid];
  __syncthreads();

  const __bf16* self = hVb + (size_t)bn * Hc;
  const __bf16* he = hEb + (size_t)bn * Kc * Hc;
  for (int i = tid; i < Kc * 16; i += blockDim.x) {
    int k = i >> 4, c = (i & 15) * 8;
    *(bf16x8*)(&sEV[k * 384 + c]) = *(const bf16x8*)(self + c);
    *(bf16x8*)(&sEV[k * 384 + 128 + c]) =
        *(const bf16x8*)(hVb + (size_t)(b * Nc + sIdx[k]) * Hc + c);
    *(bf16x8*)(&sEV[k * 384 + 256 + c]) = *(const bf16x8*)(he + (size_t)k * Hc + c);
  }
  __syncthreads();

  int nlo = lane & 15, mh = lane >> 4, col = wave * 16 + nlo;

  f32x8 acc[3] = {zero8(), zero8(), zero8()};
  gemm48(sEV, 384, W11p, 8, 12, wave, acc);
  {
    float bias = b11[col];
#pragma unroll
    for (int mt = 0; mt < 3; ++mt)
#pragma unroll
      for (int v = 0; v < 8; ++v)
        sM1[(mt * 16 + v + 8 * mh) * Hc + col] = (__bf16)gelu_f(acc[mt][v] + bias);
  }
  __syncthreads();

  f32x8 acc2[3] = {zero8(), zero8(), zero8()};
  gemm48(sM1, Hc, W12p, 8, 4, wave, acc2);
  {
    float bias = b12[col];
#pragma unroll
    for (int mt = 0; mt < 3; ++mt)
#pragma unroll
      for (int v = 0; v < 8; ++v)
        sM2[(mt * 16 + v + 8 * mh) * Hc + col] = (__bf16)gelu_f(acc2[mt][v] + bias);
  }
  __syncthreads();

  f32x8 acc3[3] = {zero8(), zero8(), zero8()};
  gemm48(sM2, Hc, W13p, 8, 4, wave, acc3);

  float* sM3 = (float*)sEV;  // reuse: 48*128*4 = 24576 <= 48*384*2 = 36864
  {
    float bias = b13[col];
#pragma unroll
    for (int mt = 0; mt < 3; ++mt)
#pragma unroll
      for (int v = 0; v < 8; ++v)
        sM3[(mt * 16 + v + 8 * mh) * Hc + col] = acc3[mt][v] + bias;
  }
  __syncthreads();

  // per-edge LayerNorm3: 6 rows per wave
  for (int r = wave * 6; r < wave * 6 + 6; ++r) {
    size_t base = ((size_t)bn * Kc + r) * Hc;
    float xv[4], s = 0.0f, s2 = 0.0f;
#pragma unroll
    for (int i = 0; i < 4; ++i) {
      int c = lane + 32 * i;
      float x = hEf[base + c] + sM3[r * Hc + c];
      xv[i] = x; s += x; s2 += x * x;
    }
#pragma unroll
    for (int off = 16; off; off >>= 1) { s += __shfl_xor(s, off); s2 += __shfl_xor(s2, off); }
    float mu = s / Hc;
    float rs = rsqrtf(s2 / Hc - mu * mu + LN_EPS);
#pragma unroll
    for (int i = 0; i < 4; ++i) {
      int c = lane + 32 * i;
      float y = lng[c] * (xv[i] - mu) * rs + lnb[c];
      hEf[base + c] = y;
      hEb[base + c] = (__bf16)y;
    }
  }
}

// ---------------------------------------------------------------------------
// Host orchestration
// ---------------------------------------------------------------------------
extern "C" void kernel_launch(void* const* d_in, const int* in_sizes, int n_in,
                              void* d_out, int out_size, void* d_ws, size_t ws_size,
                              hipStream_t stream) {
  (void)in_sizes; (void)n_in; (void)out_size; (void)ws_size;

  const float* E     = (const float*)d_in[0];
  const int*   Eidx  = (const int*)d_in[1];
  const float* mask  = (const float*)d_in[2];
  // params in setup_inputs() dict insertion order:
  const float* We_w  = (const float*)d_in[3];
  const float* We_b  = (const float*)d_in[4];
  const float* W1_w  = (const float*)d_in[5];
  const float* W1_b  = (const float*)d_in[6];
  const float* W2_w  = (const float*)d_in[7];
  const float* W2_b  = (const float*)d_in[8];
  const float* W3_w  = (const float*)d_in[9];
  const float* W3_b  = (const float*)d_in[10];
  const float* W11_w = (const float*)d_in[11];
  const float* W11_b = (const float*)d_in[12];
  const float* W12_w = (const float*)d_in[13];
  const float* W12_b = (const float*)d_in[14];
  const float* W13_w = (const float*)d_in[15];
  const float* W13_b = (const float*)d_in[16];
  const float* ffi_w = (const float*)d_in[17];
  const float* ffi_b = (const float*)d_in[18];
  const float* ffo_w = (const float*)d_in[19];
  const float* ffo_b = (const float*)d_in[20];
  const float* ln1_g = (const float*)d_in[21];
  const float* ln1_b = (const float*)d_in[22];
  const float* ln2_g = (const float*)d_in[23];
  const float* ln2_b = (const float*)d_in[24];
  const float* ln3_g = (const float*)d_in[25];
  const float* ln3_b = (const float*)d_in[26];

  const size_t BN   = (size_t)Bc * Nc;       // 4096
  const size_t BNH  = BN * Hc;               // 524288
  const size_t BNK  = BN * Kc;               // 196608
  const size_t BNKH = BNK * Hc;              // 25165824

  float* hVf = (float*)d_out;                // [B,N,H] f32 master
  float* hEf = (float*)d_out + BNH;          // [B,N,K,H] f32 master

  uint8_t* p = (uint8_t*)d_ws;
  __bf16* hEb     = (__bf16*)p; p += BNKH * 2;
  __bf16* hVb     = (__bf16*)p; p += BNH * 2;
  float*  hVmid_f = (float*)p;  p += BNH * 4;
  __bf16* hVmid_b = (__bf16*)p; p += BNH * 2;
  float*  maskAtt = (float*)p;  p += BNK * 4;
  __bf16* packWe  = (__bf16*)p; p += (size_t)16384 * 2;
  __bf16 *pW1[Lc], *pW2[Lc], *pW3[Lc], *pW11[Lc], *pW12[Lc], *pW13[Lc], *pFi[Lc], *pFo[Lc];
  for (int l = 0; l < Lc; ++l) {
    pW1[l]  = (__bf16*)p; p += (size_t)49152 * 2;
    pW2[l]  = (__bf16*)p; p += (size_t)16384 * 2;
    pW3[l]  = (__bf16*)p; p += (size_t)16384 * 2;
    pW11[l] = (__bf16*)p; p += (size_t)49152 * 2;
    pW12[l] = (__bf16*)p; p += (size_t)16384 * 2;
    pW13[l] = (__bf16*)p; p += (size_t)16384 * 2;
    pFi[l]  = (__bf16*)p; p += (size_t)65536 * 2;
    pFo[l]  = (__bf16*)p; p += (size_t)65536 * 2;
  }

  // --- repack all weights to WMMA B-fragment order (bf16) ---
  auto rp = [&](const float* W, __bf16* P, int Kd, int Nd) {
    int total = (Kd / 32) * (Nd / 16) * 32;
    repack_weight<<<(total + 255) / 256, 256, 0, stream>>>(W, P, Kd, Nd);
  };
  rp(We_w, packWe, 128, 128);
  for (int l = 0; l < Lc; ++l) {
    rp(W1_w  + (size_t)l * 384 * 128, pW1[l],  384, 128);
    rp(W2_w  + (size_t)l * 128 * 128, pW2[l],  128, 128);
    rp(W3_w  + (size_t)l * 128 * 128, pW3[l],  128, 128);
    rp(W11_w + (size_t)l * 384 * 128, pW11[l], 384, 128);
    rp(W12_w + (size_t)l * 128 * 128, pW12[l], 128, 128);
    rp(W13_w + (size_t)l * 128 * 128, pW13[l], 128, 128);
    rp(ffi_w + (size_t)l * 128 * 512, pFi[l],  128, 512);
    rp(ffo_w + (size_t)l * 512 * 128, pFo[l],  512, 128);
  }

  // --- h_E = E @ We + be ; h_V = 0 ; mask_attend ---
  edge_embed<<<(unsigned)(BNK / 128), 256, 0, stream>>>(E, packWe, We_b, hEf, hEb);
  init_hv<<<(unsigned)((BNH + 255) / 256), 256, 0, stream>>>(hVf, hVb, (int)BNH);
  make_mask_attend<<<(unsigned)((BNK + 255) / 256), 256, 0, stream>>>(mask, Eidx, maskAtt);

  // --- layers ---
  for (int l = 0; l < Lc; ++l) {
    node_message<<<(unsigned)BN, 256, 0, stream>>>(
        hVb, hVf, hEb, Eidx, maskAtt,
        pW1[l], pW2[l], pW3[l],
        W1_b + (size_t)l * Hc, W2_b + (size_t)l * Hc, W3_b + (size_t)l * Hc,
        ln1_g + (size_t)l * Hc, ln1_b + (size_t)l * Hc,
        hVmid_f, hVmid_b);

    ffn_kernel<<<(unsigned)(BN / 32), 256, 0, stream>>>(
        hVmid_f, hVmid_b, pFi[l], ffi_b + (size_t)l * 4 * Hc,
        pFo[l], ffo_b + (size_t)l * Hc,
        ln2_g + (size_t)l * Hc, ln2_b + (size_t)l * Hc, mask,
        hVf, hVb);

    edge_update<<<(unsigned)BN, 256, 0, stream>>>(
        hVb, hEf, hEb, Eidx,
        pW11[l], pW12[l], pW13[l],
        W11_b + (size_t)l * Hc, W12_b + (size_t)l * Hc, W13_b + (size_t)l * Hc,
        ln3_g + (size_t)l * Hc, ln3_b + (size_t)l * Hc);
  }
}
